// PrxteinMPNN_53068615909861
// MI455X (gfx1250) — matrix operand order repaired and checked by reading, not compiled, not run on hardware
//
#include <hip/hip_runtime.h>
#include <hip/hip_bf16.h>

// ProteinMPNN-style encoder/decoder on gfx1250 (MI455X).
// All matmuls run on v_wmma_f32_16x16x32_f16 (f16 in, f32 accumulate),
// register-blocked 32x32 per wave (4 accumulators) for 2x higher
// FLOP/byte at the load units. Residual/LayerNorm path kept in f32.

#define NN 8192
#define KNB 48
#define CC 128
#define AOUT 21
#define NK (NN * KNB)

typedef __attribute__((ext_vector_type(16))) _Float16 v16h;
typedef __attribute__((ext_vector_type(8)))  _Float16 v8h;
typedef __attribute__((ext_vector_type(8)))  float    v8f;

__device__ __forceinline__ float gelu_f(float x) {
    const float c0 = 0.7978845608028654f;   // sqrt(2/pi)
    const float c1 = 0.044715f;
    return 0.5f * x * (1.0f + tanhf(c0 * (x + c1 * x * x * x)));
}

// ---------------------------------------------------------------------------
// WMMA GEMM: Y(M x Nout) = act(X(M x Kd) @ W(Kd x Nout) + bias)
// X: f16 row-major. Wt: f16 TRANSPOSED (Nout x Kd) so each B lane reads two runs of 8
// consecutive K values at kk+hf*8 and kk+16+hf*8 (matches 16-bit B 32x16 VGPR layout).
// One wave computes a 32x32 tile (2x2 register blocking, 4 v8f accumulators);
// 8 waves per block. Requires M%32==0, Nout%32==0 (logits padded 21->32).
// ---------------------------------------------------------------------------
template <bool ACT_GELU, bool OUT_F16>
__global__ __launch_bounds__(256) void gemm_wmma(
    const _Float16* __restrict__ X, const _Float16* __restrict__ Wt,
    const float* __restrict__ bias, void* __restrict__ Y,
    int M, int Kd, int ntiles /* Nout/32 */, int ldY, int nstore)
{
    int wave = blockIdx.x * 8 + (threadIdx.x >> 5);
    int lane = threadIdx.x & 31;
    int total = (M >> 5) * ntiles;
    if (wave >= total) return;            // wave-uniform: EXEC stays all-ones
    int tm = wave / ntiles;
    int tn = wave % ntiles;

    int idx16 = lane & 15;                // A: row m   | B: col n
    int hf    = lane >> 4;                // half-wave selector

    // A lane base: rows tm*32+idx16 (+16), halves [hf*8..+7] and [+16..]
    const _Float16* a0p = X + (size_t)(tm * 32 + idx16) * Kd + hf * 8;
    const _Float16* a1p = a0p + (size_t)16 * Kd;
    // B lane base: cols tn*32+idx16 (+16), halves [hf*8..+7] and [+16..]
    const _Float16* b0p = Wt + (size_t)(tn * 32 + idx16) * Kd + hf * 8;
    const _Float16* b1p = b0p + (size_t)16 * Kd;

    v8f acc00 = {}, acc01 = {}, acc10 = {}, acc11 = {};
    for (int kk = 0; kk < Kd; kk += 32) {
        union { v16h v; v8h h[2]; } a0, a1, b0, b1;
        a0.h[0] = *(const v8h*)(a0p + kk);
        a0.h[1] = *(const v8h*)(a0p + kk + 16);
        a1.h[0] = *(const v8h*)(a1p + kk);
        a1.h[1] = *(const v8h*)(a1p + kk + 16);
        b0.h[0] = *(const v8h*)(b0p + kk);
        b0.h[1] = *(const v8h*)(b0p + kk + 16);
        b1.h[0] = *(const v8h*)(b1p + kk);
        b1.h[1] = *(const v8h*)(b1p + kk + 16);
        acc00 = __builtin_amdgcn_wmma_f32_16x16x32_f16(false, a0.v, false, b0.v, (short)0, acc00, false, false);
        acc01 = __builtin_amdgcn_wmma_f32_16x16x32_f16(false, a0.v, false, b1.v, (short)0, acc01, false, false);
        acc10 = __builtin_amdgcn_wmma_f32_16x16x32_f16(false, a1.v, false, b0.v, (short)0, acc10, false, false);
        acc11 = __builtin_amdgcn_wmma_f32_16x16x32_f16(false, a1.v, false, b1.v, (short)0, acc11, false, false);
    }

    // C/D layout per 16x16 subtile: VGPR i -> M = i + hf*8 ; N = lane&15
    auto store_tile = [&](const v8f& acc, int mt, int nt) {
        int col   = tn * 32 + nt * 16 + idx16;
        int rbase = tm * 32 + mt * 16 + hf * 8;
        if (col < nstore) {
            float bv = bias ? bias[col] : 0.0f;
#pragma unroll
            for (int i = 0; i < 8; ++i) {
                float v = acc[i] + bv;
                if (ACT_GELU) v = gelu_f(v);
                size_t o = (size_t)(rbase + i) * ldY + col;
                if (OUT_F16) ((_Float16*)Y)[o] = (_Float16)v;
                else         ((float*)Y)[o] = v;
            }
        }
    };
    store_tile(acc00, 0, 0);
    store_tile(acc01, 0, 1);
    store_tile(acc10, 1, 0);
    store_tile(acc11, 1, 1);
}

// ---------------------------------------------------------------------------
// Helper kernels
// ---------------------------------------------------------------------------

// fp32 W (Kd x Nout) -> f16 transposed Wt (Nout x Kd)
__global__ void wconv_kernel(const float* __restrict__ W, _Float16* __restrict__ Wt,
                             int Kd, int Nout)
{
    int i = blockIdx.x * blockDim.x + threadIdx.x;
    if (i >= Kd * Nout) return;
    int k = i / Nout, n = i % Nout;
    Wt[(size_t)n * Kd + k] = (_Float16)W[i];
}

__global__ void f32_to_f16_kernel(const float* __restrict__ x, _Float16* __restrict__ y,
                                  size_t n)
{
    size_t i = (size_t)blockIdx.x * blockDim.x + threadIdx.x;
    if (i < n) y[i] = (_Float16)x[i];
}

// encoder edge concat: out[e] = [hV16[n], hE16[e], hV16[nidx[e]]]  (3C)
__global__ __launch_bounds__(CC) void enc_concat_kernel(
    const _Float16* __restrict__ hV16, const _Float16* __restrict__ hE16,
    const int* __restrict__ nidx, _Float16* __restrict__ out)
{
    int e = blockIdx.x, c = threadIdx.x;
    int n = e / KNB;
    int j = nidx[e];
    _Float16* o = out + (size_t)e * (3 * CC);
    o[c]          = hV16[(size_t)n * CC + c];
    o[CC + c]     = hE16[(size_t)e * CC + c];
    o[2 * CC + c] = hV16[(size_t)j * CC + c];
}

// decoder frozen context: hEXV[e] = [0, hE16[e], hV16[nidx[e]]]  (3C)
__global__ __launch_bounds__(CC) void dec_context_kernel(
    const _Float16* __restrict__ hV16, const _Float16* __restrict__ hE16,
    const int* __restrict__ nidx, _Float16* __restrict__ out)
{
    int e = blockIdx.x, c = threadIdx.x;
    int j = nidx[e];
    _Float16* o = out + (size_t)e * (3 * CC);
    o[c]          = (_Float16)0.0f;
    o[CC + c]     = hE16[(size_t)e * CC + c];
    o[2 * CC + c] = hV16[(size_t)j * CC + c];
}

// decoder concat: out[e] = [hV16[n], hEXV[e]]  (4C)
__global__ __launch_bounds__(CC) void dec_concat_kernel(
    const _Float16* __restrict__ hV16, const _Float16* __restrict__ hEXV,
    _Float16* __restrict__ out)
{
    int e = blockIdx.x, c = threadIdx.x;
    int n = e / KNB;
    _Float16* o = out + (size_t)e * (4 * CC);
    const _Float16* x = hEXV + (size_t)e * (3 * CC);
    o[c] = hV16[(size_t)n * CC + c];
#pragma unroll
    for (int t = 0; t < 3; ++t) o[(t + 1) * CC + c] = x[t * CC + c];
}

// dh[n,c] = (1/K) * sum_k m[n,k,c] * (use_mask ? mask[n]*mask[nidx] : 1)
__global__ __launch_bounds__(CC) void edge_reduce_kernel(
    const _Float16* __restrict__ m, const float* __restrict__ mask,
    const int* __restrict__ nidx, float* __restrict__ dh, int use_mask)
{
    int n = blockIdx.x, c = threadIdx.x;
    float mn = use_mask ? mask[n] : 1.0f;
    float acc = 0.0f;
    for (int k = 0; k < KNB; ++k) {
        float w = use_mask ? (mn * mask[nidx[n * KNB + k]]) : 1.0f;
        acc += w * (float)m[((size_t)n * KNB + k) * CC + c];
    }
    dh[(size_t)n * CC + c] = acc * (1.0f / KNB);
}

// y = LN(x + resid) * g + b, optionally * mask[n]; writes f32 and f16 copies
__global__ __launch_bounds__(CC) void ln_node_kernel(
    const float* __restrict__ x, const float* __restrict__ resid,
    const float* __restrict__ g, const float* __restrict__ b,
    const float* __restrict__ mask, float* __restrict__ y32,
    _Float16* __restrict__ y16)
{
    __shared__ float red[CC];
    int n = blockIdx.x, c = threadIdx.x;
    float v = x[(size_t)n * CC + c];
    if (resid) v += resid[(size_t)n * CC + c];
    red[c] = v; __syncthreads();
#pragma unroll
    for (int s = CC / 2; s > 0; s >>= 1) { if (c < s) red[c] += red[c + s]; __syncthreads(); }
    float mu = red[0] * (1.0f / CC);
    __syncthreads();
    float d = v - mu;
    red[c] = d * d; __syncthreads();
#pragma unroll
    for (int s = CC / 2; s > 0; s >>= 1) { if (c < s) red[c] += red[c + s]; __syncthreads(); }
    float var = red[0] * (1.0f / CC);
    float o = d * rsqrtf(var + 1e-5f) * g[c] + b[c];
    if (mask) o *= mask[n];
    if (y32) y32[(size_t)n * CC + c] = o;
    if (y16) y16[(size_t)n * CC + c] = (_Float16)o;
}

// hE16[e] = LN(hE16[e] + me[e]) * g + b     (per-edge row of C)
__global__ __launch_bounds__(CC) void edge_residual_ln_kernel(
    _Float16* __restrict__ hE16, const _Float16* __restrict__ me,
    const float* __restrict__ g, const float* __restrict__ b)
{
    __shared__ float red[CC];
    int e = blockIdx.x, c = threadIdx.x;
    size_t i = (size_t)e * CC + c;
    float v = (float)hE16[i] + (float)me[i];
    red[c] = v; __syncthreads();
#pragma unroll
    for (int s = CC / 2; s > 0; s >>= 1) { if (c < s) red[c] += red[c + s]; __syncthreads(); }
    float mu = red[0] * (1.0f / CC);
    __syncthreads();
    float d = v - mu;
    red[c] = d * d; __syncthreads();
#pragma unroll
    for (int s = CC / 2; s > 0; s >>= 1) { if (c < s) red[c] += red[c + s]; __syncthreads(); }
    float var = red[0] * (1.0f / CC);
    hE16[i] = (_Float16)(d * rsqrtf(var + 1e-5f) * g[c] + b[c]);
}

// ---------------------------------------------------------------------------
// Host orchestration
// ---------------------------------------------------------------------------
static void run_gemm(hipStream_t s, const _Float16* X, const _Float16* Wt,
                     const float* bias, void* Y, int M, int Kd, int Nout,
                     int ldY, int nstore, bool act_gelu, bool out_f16)
{
    int ntiles = Nout >> 5;               // 32-col wave tiles
    int total = (M >> 5) * ntiles;
    dim3 grid((total + 7) / 8), block(256);
    if (act_gelu) {
        if (out_f16) gemm_wmma<true, true><<<grid, block, 0, s>>>(X, Wt, bias, Y, M, Kd, ntiles, ldY, nstore);
        else         gemm_wmma<true, false><<<grid, block, 0, s>>>(X, Wt, bias, Y, M, Kd, ntiles, ldY, nstore);
    } else {
        if (out_f16) gemm_wmma<false, true><<<grid, block, 0, s>>>(X, Wt, bias, Y, M, Kd, ntiles, ldY, nstore);
        else         gemm_wmma<false, false><<<grid, block, 0, s>>>(X, Wt, bias, Y, M, Kd, ntiles, ldY, nstore);
    }
}

extern "C" void kernel_launch(void* const* d_in, const int* in_sizes, int n_in,
                              void* d_out, int out_size, void* d_ws, size_t ws_size,
                              hipStream_t stream)
{
    (void)in_sizes; (void)n_in; (void)out_size; (void)ws_size;

    // JAX pytree flatten order (sorted dict keys):
    //   edge_features, mask, neighbor_indices,
    //   dec[l]: W1.b,W1.w, W2.b,W2.w, W3.b,W3.w, Wi.b,Wi.w, Wo.b,Wo.w, n1.b,n1.g, n2.b,n2.g
    //   enc[l]: W1, W11, W12, W13, W2, W3, Wi, Wo (b,w each), n1, n2, n3 (b,g each)
    //   w_out: b, w
    const float* edge_features = (const float*)d_in[0];
    const float* mask          = (const float*)d_in[1];
    const int*   nidx          = (const int*)d_in[2];
    auto P = [&](int i) -> const float* { return (const float*)d_in[i]; };
    const int DEC0 = 3;
    const int ENC0 = 3 + 3 * 14;
    const int WOUT = 3 + 3 * 14 + 3 * 22;

    // workspace sub-allocation
    char* ws = (char*)d_ws;
    size_t off = 0;
    auto alloc = [&](size_t bytes) -> char* {
        char* p = ws + off;
        off += bytes;
        off = (off + 255) & ~(size_t)255;
        return p;
    };
    _Float16* big0 = (_Float16*)alloc((size_t)NK * 4 * CC * 2);   // concat (up to 4C)
    _Float16* t1   = (_Float16*)alloc((size_t)NK * CC * 2);
    _Float16* t2   = (_Float16*)alloc((size_t)NK * CC * 2);
    _Float16* hE16 = (_Float16*)alloc((size_t)NK * CC * 2);
    _Float16* hEXV = (_Float16*)alloc((size_t)NK * 3 * CC * 2);
    float*    hV   = (float*)alloc((size_t)NN * CC * 4);
    float*    dh   = (float*)alloc((size_t)NN * CC * 4);
    _Float16* hV16 = (_Float16*)alloc((size_t)NN * CC * 2);
    _Float16* ff16 = (_Float16*)alloc((size_t)NN * 4 * CC * 2);

    // convert weight (Kd x Nout fp32) -> transposed f16 (padN x Kd)
    auto conv = [&](const float* w, int Kd, int Nout, int padN) -> _Float16* {
        _Float16* wt = (_Float16*)alloc((size_t)padN * Kd * 2);
        if (padN != Nout) hipMemsetAsync(wt, 0, (size_t)padN * Kd * 2, stream);
        int tot = Kd * Nout;
        wconv_kernel<<<(tot + 255) / 256, 256, 0, stream>>>(w, wt, Kd, Nout);
        return wt;
    };

    struct EncW {
        _Float16 *W1, *W2, *W3, *Wi, *Wo, *W11, *W12, *W13;
        const float *b1, *b2, *b3, *bi, *bo, *b11, *b12, *b13;
        const float *n1g, *n1b, *n2g, *n2b, *n3g, *n3b;
    } enc[3];
    struct DecW {
        _Float16 *W1, *W2, *W3, *Wi, *Wo;
        const float *b1, *b2, *b3, *bi, *bo;
        const float *n1g, *n1b, *n2g, *n2b;
    } dec[3];

    for (int l = 0; l < 3; ++l) {
        int B = DEC0 + l * 14;
        dec[l].b1 = P(B + 0);  dec[l].W1 = conv(P(B + 1), 4 * CC, CC, CC);
        dec[l].b2 = P(B + 2);  dec[l].W2 = conv(P(B + 3), CC, CC, CC);
        dec[l].b3 = P(B + 4);  dec[l].W3 = conv(P(B + 5), CC, CC, CC);
        dec[l].bi = P(B + 6);  dec[l].Wi = conv(P(B + 7), CC, 4 * CC, 4 * CC);
        dec[l].bo = P(B + 8);  dec[l].Wo = conv(P(B + 9), 4 * CC, CC, CC);
        dec[l].n1b = P(B + 10); dec[l].n1g = P(B + 11);
        dec[l].n2b = P(B + 12); dec[l].n2g = P(B + 13);
    }
    for (int l = 0; l < 3; ++l) {
        int B = ENC0 + l * 22;
        enc[l].b1  = P(B + 0);  enc[l].W1  = conv(P(B + 1), 3 * CC, CC, CC);
        enc[l].b11 = P(B + 2);  enc[l].W11 = conv(P(B + 3), 3 * CC, CC, CC);
        enc[l].b12 = P(B + 4);  enc[l].W12 = conv(P(B + 5), CC, CC, CC);
        enc[l].b13 = P(B + 6);  enc[l].W13 = conv(P(B + 7), CC, CC, CC);
        enc[l].b2  = P(B + 8);  enc[l].W2  = conv(P(B + 9), CC, CC, CC);
        enc[l].b3  = P(B + 10); enc[l].W3  = conv(P(B + 11), CC, CC, CC);
        enc[l].bi  = P(B + 12); enc[l].Wi  = conv(P(B + 13), CC, 4 * CC, 4 * CC);
        enc[l].bo  = P(B + 14); enc[l].Wo  = conv(P(B + 15), 4 * CC, CC, CC);
        enc[l].n1b = P(B + 16); enc[l].n1g = P(B + 17);
        enc[l].n2b = P(B + 18); enc[l].n2g = P(B + 19);
        enc[l].n3b = P(B + 20); enc[l].n3g = P(B + 21);
    }
    const float* wout_b = P(WOUT + 0);
    _Float16* wout_t = conv(P(WOUT + 1), CC, AOUT, 32);   // pad cols to 32

    // initial state: h_V = 0, h_E = edge_features (f16)
    hipMemsetAsync(hV, 0, (size_t)NN * CC * 4, stream);
    hipMemsetAsync(hV16, 0, (size_t)NN * CC * 2, stream);
    {
        size_t tot = (size_t)NK * CC;
        f32_to_f16_kernel<<<(unsigned)((tot + 255) / 256), 256, 0, stream>>>(edge_features, hE16, tot);
    }

    // ---------------- encoder ----------------
    for (int l = 0; l < 3; ++l) {
        EncW& p = enc[l];
        // message MLP on [h_i, e, h_j]
        enc_concat_kernel<<<NK, CC, 0, stream>>>(hV16, hE16, nidx, big0);
        run_gemm(stream, big0, p.W1, p.b1, t1, NK, 3 * CC, CC, CC, CC, true, true);
        run_gemm(stream, t1,   p.W2, p.b2, t2, NK, CC, CC, CC, CC, true, true);
        run_gemm(stream, t2,   p.W3, p.b3, t1, NK, CC, CC, CC, CC, false, true);
        edge_reduce_kernel<<<NN, CC, 0, stream>>>(t1, mask, nidx, dh, 1);
        ln_node_kernel<<<NN, CC, 0, stream>>>(dh, hV, p.n1g, p.n1b, nullptr, hV, hV16);
        // position-wise FFN
        run_gemm(stream, hV16, p.Wi, p.bi, ff16, NN, CC, 4 * CC, 4 * CC, 4 * CC, true, true);
        run_gemm(stream, ff16, p.Wo, p.bo, dh,   NN, 4 * CC, CC, CC, CC, false, false);
        ln_node_kernel<<<NN, CC, 0, stream>>>(dh, hV, p.n2g, p.n2b, mask, hV, hV16);
        // edge update MLP on [h_i, e, h_j] with updated nodes
        enc_concat_kernel<<<NK, CC, 0, stream>>>(hV16, hE16, nidx, big0);
        run_gemm(stream, big0, p.W11, p.b11, t1, NK, 3 * CC, CC, CC, CC, true, true);
        run_gemm(stream, t1,   p.W12, p.b12, t2, NK, CC, CC, CC, CC, true, true);
        run_gemm(stream, t2,   p.W13, p.b13, t1, NK, CC, CC, CC, CC, false, true);
        edge_residual_ln_kernel<<<NK, CC, 0, stream>>>(hE16, t1, p.n3g, p.n3b);
    }

    // frozen decoding context: [0, h_E, h_V[nidx]]
    dec_context_kernel<<<NK, CC, 0, stream>>>(hV16, hE16, nidx, hEXV);

    // ---------------- decoder ----------------
    for (int l = 0; l < 3; ++l) {
        DecW& p = dec[l];
        dec_concat_kernel<<<NK, CC, 0, stream>>>(hV16, hEXV, big0);
        run_gemm(stream, big0, p.W1, p.b1, t1, NK, 4 * CC, CC, CC, CC, true, true);
        run_gemm(stream, t1,   p.W2, p.b2, t2, NK, CC, CC, CC, CC, true, true);
        run_gemm(stream, t2,   p.W3, p.b3, t1, NK, CC, CC, CC, CC, false, true);
        edge_reduce_kernel<<<NN, CC, 0, stream>>>(t1, mask, nidx, dh, 0);
        ln_node_kernel<<<NN, CC, 0, stream>>>(dh, hV, p.n1g, p.n1b, nullptr, hV, hV16);
        run_gemm(stream, hV16, p.Wi, p.bi, ff16, NN, CC, 4 * CC, 4 * CC, 4 * CC, true, true);
        run_gemm(stream, ff16, p.Wo, p.bo, dh,   NN, 4 * CC, CC, CC, CC, false, false);
        ln_node_kernel<<<NN, CC, 0, stream>>>(dh, hV, p.n2g, p.n2b, mask, hV, hV16);
    }

    // logits = h @ w_out + b  -> d_out (N x 21 f32), padded GEMM with col clamp
    run_gemm(stream, hV16, wout_t, wout_b, d_out, NN, CC, 32, AOUT, AOUT, false, false);
}
